// GPTBlock_9199819948368
// MI455X (gfx1250) — matrix operand order repaired and checked by reading, not compiled, not run on hardware
//
#include <hip/hip_runtime.h>
#include <hip/hip_bf16.h>

// ---------------------------------------------------------------------------
// GPT block for MI455X (gfx1250): bf16 WMMA GEMMs + flash attention, fp32 acc.
// Async global->LDS copies (ASYNCcnt) pipeline the GEMM tiles.
// B=8, T=1024, S=256, C=1024, H=16, D=64, F=4096.
// ---------------------------------------------------------------------------

typedef __attribute__((ext_vector_type(16))) __bf16 v16bf;
typedef __attribute__((ext_vector_type(8)))  __bf16 v8bf;
typedef __attribute__((ext_vector_type(8)))  float  v8f;

#define Bsz 8
#define Tsz 1024
#define Ssz 256
#define Csz 1024
#define Hsz 16
#define Dsz 64
#define Fsz 4096

static __device__ __forceinline__ __bf16 f2bf(float f) {
    unsigned u = __builtin_bit_cast(unsigned, f);
    unsigned r = (u + 0x7FFFu + ((u >> 16) & 1u)) >> 16;
    unsigned short h = (unsigned short)r;
    return __builtin_bit_cast(__bf16, h);
}

// Low 32 bits of a generic shared-memory pointer are the LDS byte offset
static __device__ __forceinline__ unsigned lds_off(const void* p) {
    return (unsigned)(unsigned long long)p;
}

// 16-byte async copy global -> LDS, tracked by ASYNCcnt (bypasses VGPRs).
static __device__ __forceinline__ void async_b128(unsigned lds, const void* g) {
    asm volatile("global_load_async_to_lds_b128 %0, %1, off"
                 :: "v"(lds), "v"(g) : "memory");
}
static __device__ __forceinline__ void wait_async0() {
    asm volatile("s_wait_asynccnt 0x0" ::: "memory");
}

static __device__ __forceinline__ v16bf make_frag(const __bf16* lo, const __bf16* hi) {
    v8bf a = *(const v8bf*)lo;
    v8bf b = *(const v8bf*)hi;
    v16bf r;
#pragma unroll
    for (int i = 0; i < 8; ++i) { r[i] = a[i]; r[i + 8] = b[i]; }
    return r;
}

static __device__ __forceinline__ v8f wmma_bf16(v16bf a, v16bf b, v8f c) {
    return __builtin_amdgcn_wmma_f32_16x16x32_bf16(false, a, false, b, (short)0, c,
                                                   false, false);
}

// ---------------------------------------------------------------------------
// fp32 -> bf16 elementwise convert (weights, context)
// ---------------------------------------------------------------------------
__global__ void cvt_bf16(const float* __restrict__ X, __bf16* __restrict__ Y, long n) {
    long i = (long)blockIdx.x * blockDim.x + threadIdx.x;
    if (i < n) Y[i] = f2bf(X[i]);
}

// ---------------------------------------------------------------------------
// Fused LayerNorm (C=1024) -> bf16 rows. One 256-thread block per row.
// ---------------------------------------------------------------------------
__global__ __launch_bounds__(256)
void ln_bf16(const float* __restrict__ X, const float* __restrict__ g,
             const float* __restrict__ bta, __bf16* __restrict__ Y) {
    __shared__ float r1[256];
    __shared__ float r2[256];
    const long row = blockIdx.x;
    const float* xr = X + row * Csz;
    const int t = threadIdx.x;
    float4 xv = *(const float4*)(xr + t * 4);
    float s = xv.x + xv.y + xv.z + xv.w;
    float q = xv.x * xv.x + xv.y * xv.y + xv.z * xv.z + xv.w * xv.w;
    r1[t] = s; r2[t] = q;
    __syncthreads();
    for (int off = 128; off > 0; off >>= 1) {
        if (t < off) { r1[t] += r1[t + off]; r2[t] += r2[t + off]; }
        __syncthreads();
    }
    const float mu = r1[0] * (1.0f / Csz);
    const float var = r2[0] * (1.0f / Csz) - mu * mu;
    const float rs = rsqrtf(var + 1e-5f);
    __bf16* yr = Y + row * Csz;
#pragma unroll
    for (int i = 0; i < 4; ++i) {
        int c = t * 4 + i;
        float x = (&xv.x)[i];
        yr[c] = f2bf((x - mu) * rs * g[c] + bta[c]);
    }
}

// ---------------------------------------------------------------------------
// Tiled bf16 WMMA GEMM:  out = epilogue( A[M,K] @ B[K,N] )
// Block tile 128x128, K-step 32, 8 waves: 4 along M x 2 along N -> 8 WMMA/step.
// Double-buffered LDS + software pipeline:
//   - A tile: global_load_async_to_lds_b128 (ASYNCcnt) straight into LDS
//   - B tile: global loads issued before the WMMAs, transpose-stored after
//   - one s_wait_asynccnt + one barrier per K-step
// Epilogue: (+bias) -> (exact GELU) -> (+fp32 residual) -> fp32/bf16 stores.
// ---------------------------------------------------------------------------
#define LDA 40
#define LDB 40

__global__ __launch_bounds__(256)
void gemm_bf16_wmma(const __bf16* __restrict__ A, const __bf16* __restrict__ Bm,
                    const float* __restrict__ bias, const float* __restrict__ resid,
                    float* __restrict__ outF, __bf16* __restrict__ outB,
                    int M, int N, int K, int gelu) {
    __shared__ __bf16 sA[2][128 * LDA];
    __shared__ __bf16 sB[2][128 * LDB];

    const int t = threadIdx.x;
    const int w = t >> 5, lane = t & 31;
    const int half = lane >> 4, l16 = lane & 15;
    const int wm = w & 3, wn = w >> 2;
    const long m0 = (long)blockIdx.y * 128;
    const long n0 = (long)blockIdx.x * 128;

    // Loop-invariant copy descriptors.
    // A chunks (8 bf16 each): c = t and t+256 -> row=c>>2, col=(c&3)*8
    const int ar0 = t >> 2,          ac = (t & 3) * 8;
    const int ar1 = (t + 256) >> 2;  // same ac
    const __bf16* aS0 = A + (m0 + ar0) * (long)K + ac;
    const __bf16* aS1 = A + (m0 + ar1) * (long)K + ac;
    const unsigned aD0[2] = { lds_off(&sA[0][ar0 * LDA + ac]),
                              lds_off(&sA[1][ar0 * LDA + ac]) };
    const unsigned aD1[2] = { lds_off(&sA[0][ar1 * LDA + ac]),
                              lds_off(&sA[1][ar1 * LDA + ac]) };
    // B chunks: c = t and t+256 -> kr=c>>4 (t>>4 and +16), col=(t&15)*8
    const int br0 = t >> 4, br1 = br0 + 16, bc = (t & 15) * 8;
    const __bf16* bS0 = Bm + (long)br0 * N + n0 + bc;
    const __bf16* bS1 = Bm + (long)br1 * N + n0 + bc;

    v8f acc[2][4];
#pragma unroll
    for (int mi = 0; mi < 2; ++mi)
#pragma unroll
        for (int ni = 0; ni < 4; ++ni)
#pragma unroll
            for (int i = 0; i < 8; ++i) acc[mi][ni][i] = 0.0f;

    const int nK = K >> 5;
    v8bf vb0, vb1;

    // ---- prologue: tile 0 into buffer 0 ----
    async_b128(aD0[0], aS0);
    async_b128(aD1[0], aS1);
    vb0 = *(const v8bf*)bS0;
    vb1 = *(const v8bf*)bS1;
#pragma unroll
    for (int i = 0; i < 8; ++i) {
        sB[0][(bc + i) * LDB + br0] = vb0[i];
        sB[0][(bc + i) * LDB + br1] = vb1[i];
    }
    wait_async0();
    __syncthreads();

    for (int kt = 0; kt < nK; ++kt) {
        const int p = kt & 1;
        const bool more = (kt + 1) < nK;
        const long kn = (long)(kt + 1) * 32;
        if (more) {  // issue next tile's traffic before compute
            async_b128(aD0[p ^ 1], aS0 + kn);
            async_b128(aD1[p ^ 1], aS1 + kn);
            vb0 = *(const v8bf*)(bS0 + kn * N);
            vb1 = *(const v8bf*)(bS1 + kn * N);
        }

        v16bf af[2], bfr[4];
#pragma unroll
        for (int mi = 0; mi < 2; ++mi) {
            const int Mr = wm * 32 + mi * 16 + l16;
            af[mi] = make_frag(&sA[p][Mr * LDA + half * 8],
                               &sA[p][Mr * LDA + half * 8 + 16]);
        }
#pragma unroll
        for (int ni = 0; ni < 4; ++ni) {
            const int Nr = wn * 64 + ni * 16 + l16;
            bfr[ni] = make_frag(&sB[p][Nr * LDB + half * 8],
                                &sB[p][Nr * LDB + half * 8 + 16]);
        }
#pragma unroll
        for (int mi = 0; mi < 2; ++mi)
#pragma unroll
            for (int ni = 0; ni < 4; ++ni)
                acc[mi][ni] = wmma_bf16(af[mi], bfr[ni], acc[mi][ni]);

        if (more) {  // drain next tile: transpose-store B, join async A
#pragma unroll
            for (int i = 0; i < 8; ++i) {
                sB[p ^ 1][(bc + i) * LDB + br0] = vb0[i];
                sB[p ^ 1][(bc + i) * LDB + br1] = vb1[i];
            }
            wait_async0();
            __syncthreads();
        }
    }

    // Epilogue. C/D layout: VGPR r -> row (r + 8*half), col = l16 in 16x16 tile.
#pragma unroll
    for (int mi = 0; mi < 2; ++mi) {
#pragma unroll
        for (int ni = 0; ni < 4; ++ni) {
            const long gc = n0 + wn * 64 + ni * 16 + l16;
            const float bv = bias ? bias[gc] : 0.0f;
#pragma unroll
            for (int r = 0; r < 8; ++r) {
                const long gr = m0 + wm * 32 + mi * 16 + r + half * 8;
                float v = acc[mi][ni][r] + bv;
                if (gelu) v = 0.5f * v * (1.0f + erff(v * 0.70710678118f));
                if (resid) v += resid[gr * N + gc];
                if (outF) outF[gr * N + gc] = v;
                if (outB) outB[gr * N + gc] = f2bf(v);
            }
        }
    }
}

// ---------------------------------------------------------------------------
// Flash attention (bf16 WMMA, fp32 online softmax). 4 waves, 64 q-rows per WG
// (16 per wave), 32-key blocks. Q and K tiles copied via async-to-LDS;
// V goes through VGPRs for the in-register transpose. Per key block per wave:
// 4 WMMA (Q.K^T, K=64 chained) + 4 WMMA (P.V). Output merged [B*T, C] bf16.
// ---------------------------------------------------------------------------
__global__ __launch_bounds__(128)
void attn_wmma(const __bf16* __restrict__ Q, long qStride, long qCol0,
               const __bf16* __restrict__ Kp, long kStride, long kCol0,
               const __bf16* __restrict__ Vp, long vStride, long vCol0,
               __bf16* __restrict__ O, int Lk, int causal, float scale) {
    __shared__ __bf16 sQ[64 * 72];
    __shared__ __bf16 sK[32 * 72];
    __shared__ __bf16 sVT[64 * 40];   // transposed: sVT[d][key]
    __shared__ __bf16 sP[4][16 * 40]; // per-wave P tile

    const int t = threadIdx.x;
    const int w = t >> 5, lane = t & 31;
    const int half = lane >> 4, l16 = lane & 15;
    const int b = blockIdx.y / Hsz, h = blockIdx.y % Hsz;
    const int q0 = blockIdx.x * 64;
    const long qc = qCol0 + (long)h * Dsz;
    const long kc = kCol0 + (long)h * Dsz;
    const long vc = vCol0 + (long)h * Dsz;

    // Q tile 64x64 -> LDS via async copies
#pragma unroll
    for (int it = 0; it < 4; ++it) {
        int c = t + 128 * it;
        int row = c >> 3, cc = (c & 7) * 8;
        async_b128(lds_off(&sQ[row * 72 + cc]),
                   Q + ((long)b * Tsz + q0 + row) * qStride + qc + cc);
    }
    wait_async0();
    __syncthreads();

    v16bf qa0, qa1;
    {
        const int Mr = w * 16 + l16;
        qa0 = make_frag(&sQ[Mr * 72 + half * 8],      &sQ[Mr * 72 + half * 8 + 16]);
        qa1 = make_frag(&sQ[Mr * 72 + 32 + half * 8], &sQ[Mr * 72 + 32 + half * 8 + 16]);
    }

    float mrow[8], lrow[8];
    v8f o[4];
#pragma unroll
    for (int r = 0; r < 8; ++r) { mrow[r] = -1e30f; lrow[r] = 0.0f; }
#pragma unroll
    for (int nt = 0; nt < 4; ++nt)
#pragma unroll
        for (int i = 0; i < 8; ++i) o[nt][i] = 0.0f;

    int kEnd = causal ? (q0 + 64) : Lk;
    if (kEnd > Lk) kEnd = Lk;

    const int lrow8 = t >> 3, lcc = (t & 7) * 8;  // per-thread K/V chunk 0
    for (int j0 = 0; j0 < kEnd; j0 += 32) {
        __syncthreads();  // protect sK/sVT from readers of previous block
#pragma unroll
        for (int it = 0; it < 2; ++it) {
            int row = lrow8 + it * 16, cc = lcc;
            async_b128(lds_off(&sK[row * 72 + cc]),
                       Kp + ((long)b * Lk + j0 + row) * kStride + kc + cc);
            v8bf vv = *(const v8bf*)(Vp + ((long)b * Lk + j0 + row) * vStride + vc + cc);
#pragma unroll
            for (int i = 0; i < 8; ++i) sVT[(cc + i) * 40 + row] = vv[i];
        }
        wait_async0();
        __syncthreads();

        // S = Q(16x64) . K^T(64x32): two 16x16 tiles, K reduced in 2 chained steps
        v8f s[2];
#pragma unroll
        for (int nt = 0; nt < 2; ++nt) {
            const int Nr = nt * 16 + l16;  // key-local column
            v16bf kb0 = make_frag(&sK[Nr * 72 + half * 8],      &sK[Nr * 72 + half * 8 + 16]);
            v16bf kb1 = make_frag(&sK[Nr * 72 + 32 + half * 8], &sK[Nr * 72 + 32 + half * 8 + 16]);
            v8f z;
#pragma unroll
            for (int i = 0; i < 8; ++i) z[i] = 0.0f;
            z = wmma_bf16(qa0, kb0, z);
            z = wmma_bf16(qa1, kb1, z);
            s[nt] = z;
        }

        // Online softmax per row (rows replicated across the 16 lanes of a half).
#pragma unroll
        for (int r = 0; r < 8; ++r) {
            float x0 = s[0][r] * scale;
            float x1 = s[1][r] * scale;
            const int qg = q0 + w * 16 + r + half * 8;
            if (causal) {
                if (j0 + l16 > qg)      x0 = -1e30f;
                if (j0 + 16 + l16 > qg) x1 = -1e30f;
            }
            float mx = fmaxf(x0, x1);
#pragma unroll
            for (int dd = 1; dd < 16; dd <<= 1) mx = fmaxf(mx, __shfl_xor(mx, dd, 32));
            const float mn = fmaxf(mrow[r], mx);
            const float corr = __expf(mrow[r] - mn);
            const float p0 = __expf(x0 - mn);
            const float p1 = __expf(x1 - mn);
            float ps = p0 + p1;
#pragma unroll
            for (int dd = 1; dd < 16; dd <<= 1) ps += __shfl_xor(ps, dd, 32);
            lrow[r] = lrow[r] * corr + ps;
            mrow[r] = mn;
#pragma unroll
            for (int nt = 0; nt < 4; ++nt) o[nt][r] *= corr;
            const int pr = r + half * 8;  // C-layout row -> A-layout staging via LDS
            sP[w][pr * 40 + l16]      = f2bf(p0);
            sP[w][pr * 40 + 16 + l16] = f2bf(p1);
        }

        // O += P(16x32) . V(32x64). Same-wave LDS store->load is in-order.
        v16bf pa = make_frag(&sP[w][l16 * 40 + half * 8],
                             &sP[w][l16 * 40 + half * 8 + 16]);
#pragma unroll
        for (int nt = 0; nt < 4; ++nt) {
            v16bf vb = make_frag(&sVT[(nt * 16 + l16) * 40 + half * 8],
                                 &sVT[(nt * 16 + l16) * 40 + half * 8 + 16]);
            o[nt] = wmma_bf16(pa, vb, o[nt]);
        }
    }

    // Normalize and write merged [B*T, C] bf16.
#pragma unroll
    for (int nt = 0; nt < 4; ++nt) {
#pragma unroll
        for (int r = 0; r < 8; ++r) {
            const long row = (long)b * Tsz + q0 + w * 16 + r + half * 8;
            const long col = (long)h * Dsz + nt * 16 + l16;
            O[row * Csz + col] = f2bf(o[nt][r] / lrow[r]);
        }
    }
}

// ---------------------------------------------------------------------------
// Host-side orchestration
// ---------------------------------------------------------------------------
extern "C" void kernel_launch(void* const* d_in, const int* in_sizes, int n_in,
                              void* d_out, int out_size, void* d_ws, size_t ws_size,
                              hipStream_t stream) {
    (void)in_sizes; (void)n_in; (void)out_size; (void)ws_size;

    const float* x       = (const float*)d_in[0];
    const float* ctx     = (const float*)d_in[1];
    const float* ln1_g   = (const float*)d_in[2];
    const float* ln1_b   = (const float*)d_in[3];
    const float* ln2_g   = (const float*)d_in[4];
    const float* ln2_b   = (const float*)d_in[5];
    const float* ln3_g   = (const float*)d_in[6];
    const float* ln3_b   = (const float*)d_in[7];
    const float* W_qkv   = (const float*)d_in[8];
    const float* W_sproj = (const float*)d_in[9];
    const float* b_sproj = (const float*)d_in[10];
    const float* W_q     = (const float*)d_in[11];
    const float* W_kv    = (const float*)d_in[12];
    const float* W_cproj = (const float*)d_in[13];
    const float* b_cproj = (const float*)d_in[14];
    const float* W_ff1   = (const float*)d_in[15];
    const float* b_ff1   = (const float*)d_in[16];
    const float* W_ff2   = (const float*)d_in[17];
    const float* b_ff2   = (const float*)d_in[18];
    float* out = (float*)d_out;

    const long NT = (long)Bsz * Tsz;   // 8192 activation rows
    const long NS = (long)Bsz * Ssz;   // 2048 context rows

    // Workspace layout (256B aligned)
    char* ws = (char*)d_ws;
    size_t off = 0;
    auto alloc = [&](size_t bytes) -> char* {
        char* p = ws + off;
        off = (off + bytes + 255) & ~(size_t)255;
        return p;
    };
    __bf16* wqkv   = (__bf16*)alloc((size_t)Csz * 3 * Csz * 2);
    __bf16* wsproj = (__bf16*)alloc((size_t)Csz * Csz * 2);
    __bf16* wq     = (__bf16*)alloc((size_t)Csz * Csz * 2);
    __bf16* wkv    = (__bf16*)alloc((size_t)Csz * 2 * Csz * 2);
    __bf16* wcproj = (__bf16*)alloc((size_t)Csz * Csz * 2);
    __bf16* wff1   = (__bf16*)alloc((size_t)Csz * Fsz * 2);
    __bf16* wff2   = (__bf16*)alloc((size_t)Fsz * Csz * 2);
    float*  xbuf   = (float*)alloc((size_t)NT * Csz * 4);
    __bf16* hbf    = (__bf16*)alloc((size_t)NT * Csz * 2);
    __bf16* R      = (__bf16*)alloc((size_t)64 * 1024 * 1024);  // multi-purpose
    __bf16* ctxbf  = (__bf16*)alloc((size_t)NS * Csz * 2);

    __bf16* attn   = R + (size_t)48 * 1024 * 1024 / 2;  // [NT, C] bf16
    __bf16* kvx    = R + (size_t)16 * 1024 * 1024 / 2;  // cross kv [NS, 2C] bf16

    auto cvt = [&](const float* src, __bf16* dst, long n) {
        cvt_bf16<<<dim3((unsigned)((n + 255) / 256)), dim3(256), 0, stream>>>(src, dst, n);
    };
    auto gemm = [&](const __bf16* A, const __bf16* Bm, const float* bias,
                    const float* resid, float* oF, __bf16* oB,
                    int M, int N, int K, int gelu) {
        gemm_bf16_wmma<<<dim3((unsigned)(N / 128), (unsigned)(M / 128)),
                         dim3(256), 0, stream>>>(A, Bm, bias, resid, oF, oB, M, N, K, gelu);
    };

    // 0) one-time bf16 conversions
    cvt(W_qkv,   wqkv,   (long)Csz * 3 * Csz);
    cvt(W_sproj, wsproj, (long)Csz * Csz);
    cvt(W_q,     wq,     (long)Csz * Csz);
    cvt(W_kv,    wkv,    (long)Csz * 2 * Csz);
    cvt(W_cproj, wcproj, (long)Csz * Csz);
    cvt(W_ff1,   wff1,   (long)Csz * Fsz);
    cvt(W_ff2,   wff2,   (long)Fsz * Csz);
    cvt(ctx,     ctxbf,  (long)NS * Csz);

    const float scale = 0.125f;  // 1/sqrt(64)

    // --- causal self-attention ---
    ln_bf16<<<dim3((unsigned)NT), dim3(256), 0, stream>>>(x, ln1_g, ln1_b, hbf);
    gemm(hbf, wqkv, nullptr, nullptr, nullptr, R, (int)NT, 3 * Csz, Csz, 0);  // qkv
    attn_wmma<<<dim3(Tsz / 64, Bsz * Hsz), dim3(128), 0, stream>>>(
        R, 3 * Csz, 0, R, 3 * Csz, Csz, R, 3 * Csz, 2 * Csz, attn, Tsz, 1, scale);
    gemm(attn, wsproj, b_sproj, x, xbuf, nullptr, (int)NT, Csz, Csz, 0);      // + resid x

    // --- cross-attention ---
    ln_bf16<<<dim3((unsigned)NT), dim3(256), 0, stream>>>(xbuf, ln2_g, ln2_b, hbf);
    gemm(hbf, wq, nullptr, nullptr, nullptr, R, (int)NT, Csz, Csz, 0);        // q
    gemm(ctxbf, wkv, nullptr, nullptr, nullptr, kvx, (int)NS, 2 * Csz, Csz, 0); // kv
    attn_wmma<<<dim3(Tsz / 64, Bsz * Hsz), dim3(128), 0, stream>>>(
        R, Csz, 0, kvx, 2 * Csz, 0, kvx, 2 * Csz, Csz, attn, Ssz, 0, scale);
    gemm(attn, wcproj, b_cproj, xbuf, xbuf, nullptr, (int)NT, Csz, Csz, 0);   // in-place resid

    // --- FFN ---
    ln_bf16<<<dim3((unsigned)NT), dim3(256), 0, stream>>>(xbuf, ln3_g, ln3_b, hbf);
    gemm(hbf, wff1, b_ff1, nullptr, nullptr, R, (int)NT, Fsz, Csz, 1);        // exact GELU
    gemm(R, wff2, b_ff2, xbuf, out, nullptr, (int)NT, Csz, Fsz, 0);           // final out
}
